// StandardDelta_35588099014927
// MI455X (gfx1250) — compile-verified, matrix operand order checked by reading, not run on hardware
//
#include <hip/hip_runtime.h>
#include <hip/hip_bf16.h>

#define BB 64
#define LL 512
#define HH 128
#define VV 32000

typedef __attribute__((ext_vector_type(16))) _Float16 v16h;
typedef __attribute__((ext_vector_type(8)))  float    v8f;
typedef __attribute__((ext_vector_type(2)))  float    v2f;

// LDS byte offset of a __shared__ object: generic LDS address low 32 bits.
__device__ __forceinline__ unsigned lds_off(const void* p) {
    return (unsigned)(unsigned long long)p;
}
// CDNA5 async global -> LDS copy, 16B per lane (one wave moves 512B).
__device__ __forceinline__ void async_copy_b128(unsigned lds_addr, const float* gaddr) {
    asm volatile("global_load_async_to_lds_b128 %0, %1, off"
                 :: "v"(lds_addr), "v"(gaddr) : "memory");
}
__device__ __forceinline__ void async_wait0() {
    asm volatile("s_wait_asynccnt 0" ::: "memory");
}

// ---------------- embedding gather + f16 copy ----------------
__global__ void k_embed(const int* __restrict__ seq, const float* __restrict__ ew,
                        float* __restrict__ h0f, _Float16* __restrict__ h0h) {
    int row = blockIdx.x;          // B*L rows
    int t   = threadIdx.x;         // 128 threads = H
    int tok = seq[row];
    float v = ew[(size_t)tok * HH + t];
    h0f[(size_t)row * HH + t] = v;
    h0h[(size_t)row * HH + t] = (_Float16)v;
}

// ---------------- f32 -> f16 convert ----------------
__global__ void k_cvt(const float* __restrict__ src, _Float16* __restrict__ dst, int n) {
    int i = blockIdx.x * 256 + threadIdx.x;
    if (i < n) dst[i] = (_Float16)src[i];
}

// ---------------- f16 WMMA GEMM: C = A(MxK) * W(NxK)^T (+bias)(+relu)(+resid) ----------------
// one wave per 64x16 output block (4 m-tiles share each B fetch), 8 waves per block.
// All 4 A tiles are loaded into distinct live values BEFORE the WMMA sequence so the
// compiler keeps 4 independent load->wmma chains in flight (partial s_wait_loadcnt).
// Variants are compile-time so the epilogue is branch-free straight-line stores.
template <int RELU, int HAS_BIAS, int HAS_RESID, int OUT_H, int OUT_F>
__global__ __launch_bounds__(256)
void k_gemm_f16(const _Float16* __restrict__ A, const _Float16* __restrict__ W,
                const float* __restrict__ bias,
                _Float16* __restrict__ outh, float* __restrict__ outf,
                const float* __restrict__ resid,
                int M, int N, int K) {
    int wave    = blockIdx.x * (blockDim.x >> 5) + (threadIdx.x >> 5);
    int tiles_n = N >> 4;
    int total   = (M >> 6) * tiles_n;          // M blocked by 64
    if (wave >= total) return;                 // uniform per wave: EXEC stays all-ones
    int m0 = (wave / tiles_n) << 6;
    int n0 = (wave % tiles_n) << 4;
    int l = threadIdx.x & 31, half = l >> 4, lane16 = l & 15;

    v8f c[4];
    float bv = HAS_BIAS ? bias[n0 + lane16] : 0.0f;
#pragma unroll
    for (int mt = 0; mt < 4; ++mt)
#pragma unroll
        for (int i = 0; i < 8; ++i) c[mt][i] = bv;

    for (int k0 = 0; k0 < K; k0 += 32) {
        const _Float16* wp = W + (size_t)(n0 + lane16) * K + k0 + half * 8;
        v16h b;
#pragma unroll
        for (int i = 0; i < 8; ++i) { b[i] = wp[i]; b[i + 8] = wp[i + 16]; }
        v16h a[4];
#pragma unroll
        for (int mt = 0; mt < 4; ++mt) {
            const _Float16* ap = A + (size_t)(m0 + mt * 16 + lane16) * K + k0 + half * 8;
#pragma unroll
            for (int i = 0; i < 8; ++i) { a[mt][i] = ap[i]; a[mt][i + 8] = ap[i + 16]; }
        }
#pragma unroll
        for (int mt = 0; mt < 4; ++mt)
            c[mt] = __builtin_amdgcn_wmma_f32_16x16x32_f16(false, a[mt], false, b,
                                                           (short)0, c[mt], false, false);
    }

#pragma unroll
    for (int mt = 0; mt < 4; ++mt)
#pragma unroll
        for (int i = 0; i < 8; ++i) {
            int m = m0 + mt * 16 + i + half * 8;
            int n = n0 + lane16;
            float v = c[mt][i];
            if (RELU)      v = v > 0.0f ? v : 0.0f;
            if (HAS_RESID) v += resid[(size_t)m * N + n];
            if (OUT_H)     outh[(size_t)m * N + n] = (_Float16)v;
            if (OUT_F)     outf[(size_t)m * N + n] = v;
        }
}

// ---------------- f32 WMMA GEMM (16x16x4): out = A(MxK) * W(NxK)^T + bias ----------------
// one wave per 64x16 output block: W (the big streamed operand) is read exactly once.
__global__ __launch_bounds__(256)
void k_gemm_f32(const float* __restrict__ A, const float* __restrict__ W,
                const float* __restrict__ bias, float* __restrict__ out,
                int M, int N, int K) {
    int wave    = blockIdx.x * (blockDim.x >> 5) + (threadIdx.x >> 5);
    int tiles_n = N >> 4;
    int total   = (M >> 6) * tiles_n;
    if (wave >= total) return;
    int m0 = (wave / tiles_n) << 6;
    int n0 = (wave % tiles_n) << 4;
    int l = threadIdx.x & 31, half = l >> 4, lane16 = l & 15;

    v8f c[4];
    float bv = bias[n0 + lane16];
#pragma unroll
    for (int mt = 0; mt < 4; ++mt)
#pragma unroll
        for (int i = 0; i < 8; ++i) c[mt][i] = bv;

    for (int k0 = 0; k0 < K; k0 += 4) {
        const float* wp = W + (size_t)(n0 + lane16) * K + k0 + half * 2;
        __builtin_prefetch(wp + 32);           // global_prefetch_b8 on the W stream
        v2f b; b[0] = wp[0]; b[1] = wp[1];
        v2f a[4];
#pragma unroll
        for (int mt = 0; mt < 4; ++mt) {
            const float* ap = A + (size_t)(m0 + mt * 16 + lane16) * K + k0 + half * 2;
            a[mt][0] = ap[0]; a[mt][1] = ap[1];
        }
#pragma unroll
        for (int mt = 0; mt < 4; ++mt)
            c[mt] = __builtin_amdgcn_wmma_f32_16x16x4_f32(false, a[mt], false, b,
                                                          (short)0, c[mt], false, false);
    }
#pragma unroll
    for (int mt = 0; mt < 4; ++mt)
#pragma unroll
        for (int i = 0; i < 8; ++i)
            out[(size_t)(m0 + mt * 16 + i + half * 8) * N + (n0 + lane16)] = c[mt][i];
}

// ---------------- layernorm over H=128, writes f16 ----------------
__global__ void k_ln(const float* __restrict__ x, const float* __restrict__ g,
                     const float* __restrict__ b, _Float16* __restrict__ hh) {
    __shared__ float red[HH];
    int row = blockIdx.x, t = threadIdx.x;
    float v = x[(size_t)row * HH + t];
    red[t] = v; __syncthreads();
    for (int s = 64; s > 0; s >>= 1) { if (t < s) red[t] += red[t + s]; __syncthreads(); }
    float mu = red[0] * (1.0f / HH);
    __syncthreads();
    float d = v - mu;
    red[t] = d * d; __syncthreads();
    for (int s = 64; s > 0; s >>= 1) { if (t < s) red[t] += red[t + s]; __syncthreads(); }
    float var = red[0] * (1.0f / HH);
    float y = d * rsqrtf(var + 1e-5f) * g[t] + b[t];
    hh[(size_t)row * HH + t] = (_Float16)y;
}

// ---------------- L2-normalize key rows ----------------
__global__ void k_normkeys(const float* __restrict__ ks, float* __restrict__ kn) {
    __shared__ float red[HH];
    int row = blockIdx.x, t = threadIdx.x;
    float v = ks[(size_t)row * HH + t];
    red[t] = v * v; __syncthreads();
    for (int s = 64; s > 0; s >>= 1) { if (t < s) red[t] += red[t + s]; __syncthreads(); }
    float nrm = sqrtf(red[0]);
    nrm = nrm < 1e-12f ? 1e-12f : nrm;
    kn[(size_t)row * HH + t] = v / nrm;
}

// ---------------- delta-rule scan ----------------
// Rows of M are independent; split B x 2 blocks. k/kn rows for step t+1 are
// staged into LDS with async global->LDS copies (ASYNCcnt) while step t computes.
#define SCAN_SPLIT 2
#define ROWS_PB (HH / SCAN_SPLIT)   // 64 rows of M per block
__global__ __launch_bounds__(256)
void k_scan(const float* __restrict__ ks, const float* __restrict__ kn,
            float* __restrict__ readv) {
    __shared__ float Msh[ROWS_PB * (HH + 1)];  // pad kills bank conflicts
    __shared__ float kbuf[2][HH];
    __shared__ float knbuf[2][HH];
    __shared__ float dv[ROWS_PB];
    int b     = blockIdx.x / SCAN_SPLIT;
    int rbase = (blockIdx.x % SCAN_SPLIT) * ROWS_PB;
    int t     = threadIdx.x;                   // 256 threads = 8 waves
    int wv    = t >> 5, lane = t & 31;

    for (int i = t; i < ROWS_PB * (HH + 1); i += 256) Msh[i] = 0.0f;

    // preload step 0 into buffer 0 (wave 0, one 512B row per instruction)
    if (wv == 0) {
        const float* kr  = ks + (size_t)b * LL * HH;
        const float* knr = kn + (size_t)b * LL * HH;
        async_copy_b128(lds_off(&kbuf[0][lane * 4]),  kr  + lane * 4);
        async_copy_b128(lds_off(&knbuf[0][lane * 4]), knr + lane * 4);
        async_wait0();
    }
    __syncthreads();

    for (int step = 0; step < LL - 1; ++step) {
        int cur = step & 1, nxt = cur ^ 1;
        // issue async stage of step+1 (or of q row after the last update step)
        if (wv == 0) {
            int s2  = step + 1;
            int sk  = s2;                          // s2 == LL-1 stages the q row
            int skn = (s2 < LL - 1) ? s2 : (LL - 2);
            const float* kr  = ks + ((size_t)b * LL + sk)  * HH;
            const float* knr = kn + ((size_t)b * LL + skn) * HH;
            async_copy_b128(lds_off(&kbuf[nxt][lane * 4]),  kr  + lane * 4);
            async_copy_b128(lds_off(&knbuf[nxt][lane * 4]), knr + lane * 4);
        }
        // phase 1: vp = M @ kn_t ; d = k_t - vp
        if (t < ROWS_PB) {
            float acc = 0.0f;
#pragma unroll 8
            for (int j = 0; j < HH; ++j) acc += Msh[t * (HH + 1) + j] * knbuf[cur][j];
            dv[t] = kbuf[cur][rbase + t] - acc;
        }
        __syncthreads();
        // phase 2: M += d (outer) kn_t, 4 threads per row
        int r  = t >> 2;
        int cb = (t & 3) * 32;
        float d = dv[r];
#pragma unroll 8
        for (int j = 0; j < 32; ++j) Msh[r * (HH + 1) + cb + j] += d * knbuf[cur][cb + j];
        if (wv == 0) async_wait0();                // drain stage for step+1
        __syncthreads();
    }

    // read = M_final @ q  (q row was staged into kbuf[(LL-1)&1])
    const float* q = kbuf[(LL - 1) & 1];
    if (t < ROWS_PB) {
        float acc = 0.0f;
#pragma unroll 8
        for (int j = 0; j < HH; ++j) acc += Msh[t * (HH + 1) + j] * q[j];
        readv[(size_t)b * HH + rbase + t] = acc;
    }
}

extern "C" void kernel_launch(void* const* d_in, const int* in_sizes, int n_in,
                              void* d_out, int out_size, void* d_ws, size_t ws_size,
                              hipStream_t stream) {
    const int*   seq     = (const int*)  d_in[0];
    const float* embed_w = (const float*)d_in[1];
    const float* ff_w1   = (const float*)d_in[2];
    const float* ff_b1   = (const float*)d_in[3];
    const float* ff_w2   = (const float*)d_in[4];
    const float* ff_b2   = (const float*)d_in[5];
    const float* ln_g    = (const float*)d_in[6];
    const float* ln_b    = (const float*)d_in[7];
    const float* kp_w    = (const float*)d_in[8];
    const float* rp_w    = (const float*)d_in[9];
    const float* rp_b    = (const float*)d_in[10];
    const float* out_w   = (const float*)d_in[11];
    const float* out_b   = (const float*)d_in[12];
    float* out = (float*)d_out;

    const size_t ROWS = (size_t)BB * LL;       // 32768
    char* base = (char*)d_ws;
    size_t off = 0;
    auto alloc = [&](size_t bytes) -> char* {
        char* p = base + off;
        off += (bytes + 255) & ~(size_t)255;
        return p;
    };
    float*     h0f   = (float*)    alloc(ROWS * HH * 4);       // later reused as ks
    float*     xf    = (float*)    alloc(ROWS * HH * 4);       // later reused as kn
    _Float16*  ff1h  = (_Float16*) alloc(ROWS * 2 * HH * 2);
    _Float16*  h0h   = (_Float16*) alloc(ROWS * HH * 2);
    _Float16*  hh    = (_Float16*) alloc(ROWS * HH * 2);
    _Float16*  w1h   = (_Float16*) alloc(2 * HH * HH * 2);
    _Float16*  w2h   = (_Float16*) alloc(HH * 2 * HH * 2);
    _Float16*  kph   = (_Float16*) alloc(HH * HH * 2);
    float*     readv = (float*)    alloc(BB * HH * 4);
    float*     tmpf  = (float*)    alloc(BB * HH * 4);
    float*     ksf   = h0f;   // reuse: h0f dead after FF2 residual add
    float*     knf   = xf;    // reuse: xf dead after layernorm

    // 1) embedding gather (+f16 copy)
    k_embed<<<(int)ROWS, HH, 0, stream>>>(seq, embed_w, h0f, h0h);

    // 2) weight conversions to f16
    k_cvt<<<(2 * HH * HH + 255) / 256, 256, 0, stream>>>(ff_w1, w1h, 2 * HH * HH);
    k_cvt<<<(2 * HH * HH + 255) / 256, 256, 0, stream>>>(ff_w2, w2h, 2 * HH * HH);
    k_cvt<<<(HH * HH + 255) / 256, 256, 0, stream>>>(kp_w, kph, HH * HH);

    // 3) FF1: relu(h0 @ W1^T + b1) -> f16   M=32768 N=256 K=128
    {
        int waves = (int)(ROWS / 64) * (2 * HH / 16);   // 8192
        k_gemm_f16<1, 1, 0, 1, 0><<<waves / 8, 256, 0, stream>>>(
            h0h, w1h, ff_b1, ff1h, nullptr, nullptr, (int)ROWS, 2 * HH, HH);
    }
    // 4) FF2 + residual: x = h0 + (ff1 @ W2^T + b2)    M=32768 N=128 K=256
    {
        int waves = (int)(ROWS / 64) * (HH / 16);       // 4096
        k_gemm_f16<0, 1, 1, 0, 1><<<waves / 8, 256, 0, stream>>>(
            ff1h, w2h, ff_b2, nullptr, xf, h0f, (int)ROWS, HH, 2 * HH);
    }
    // 5) layernorm -> hh (f16)
    k_ln<<<(int)ROWS, HH, 0, stream>>>(xf, ln_g, ln_b, hh);

    // 6) key projection: ks = h @ kp_w^T (all rows; last row is q)  M=32768 N=128 K=128
    {
        int waves = (int)(ROWS / 64) * (HH / 16);
        k_gemm_f16<0, 0, 0, 0, 1><<<waves / 8, 256, 0, stream>>>(
            hh, kph, nullptr, nullptr, ksf, nullptr, (int)ROWS, HH, HH);
    }
    // 7) normalize keys
    k_normkeys<<<(int)ROWS, HH, 0, stream>>>(ksf, knf);

    // 8) delta-rule scan (serial in L, parallel over B x H rows), read = M_final @ q
    k_scan<<<BB * SCAN_SPLIT, 256, 0, stream>>>(ksf, knf, readv);

    // 9) tmp = read @ rp_w^T + rp_b    M=64 N=128 K=128 (f32 WMMA)
    k_gemm_f32<<<1, 256, 0, stream>>>(readv, rp_w, rp_b, tmpf, BB, HH, HH);

    // 10) out = tmp @ out_w^T + out_b  M=64 N=32000 K=128 (f32 WMMA, streams out_w once)
    {
        int waves = (BB / 64) * (VV / 16);              // 2000
        k_gemm_f32<<<waves / 8, 256, 0, stream>>>(tmpf, out_w, out_b, out, BB, VV, HH);
    }
}
